// TemporalAttention_377957122432
// MI455X (gfx1250) — matrix operand order repaired and checked by reading, not compiled, not run on hardware
//
#include <hip/hip_runtime.h>

// ---------------------------------------------------------------------------
// TemporalAttention for MI455X (gfx1250), wave32, WMMA bf16 16x16x32 path.
// B=256, S=100, D=512, H=8, HD=64.
// d_out = [out: 256*100*512 f32][attn: 256*8*100*100 f32]
// ---------------------------------------------------------------------------

typedef __attribute__((ext_vector_type(16))) __bf16          v16bf;
typedef __attribute__((ext_vector_type(16))) unsigned short  v16us;
typedef __attribute__((ext_vector_type(8)))  float           v8f;

#define NB  256          // B
#define NS  100          // S
#define ND  512          // D
#define NH  8            // H
#define NHD 64           // head dim
#define NROWS (NB * NS)  // 25600
#define SP  112          // S padded to 7*16
#define KP  128          // S padded to 4*32 (K dim of ctx GEMM)

// f32 -> bf16 via native conversion (lowers to v_cvt on gfx1250)
__device__ __forceinline__ unsigned short f2bf(float f) {
    __bf16 h = (__bf16)f;
    return __builtin_bit_cast(unsigned short, h);
}

// A-fragment (16x32 bf16) per ISA 7.12.2: lane L -> row M0+(L&15);
// VGPR v<4 holds K = 2v,2v+1 (+8 if lane>=16); v>=4 holds K = 16+2(v-4) (+8).
__device__ __forceinline__ v16bf load_a_bf16(const unsigned short* src, int ld,
                                             int M0, int kk) {
    int lane = threadIdx.x & 31;
    int g = lane >> 4;
    const unsigned short* r = src + (size_t)(M0 + (lane & 15)) * ld + kk;
    v16us u;
#pragma unroll
    for (int v = 0; v < 8; ++v) {
        int k0 = (v < 4 ? 2 * v : 16 + 2 * (v - 4)) + 8 * g;
        u[2 * v]     = r[k0];
        u[2 * v + 1] = r[k0 + 1];
    }
    return __builtin_bit_cast(v16bf, u);
}

// B-fragment (32x16 bf16), B(k,n) = src[n*ld + k] (n-major storage, k contiguous).
// Lane L -> column N0+(L&15); halves hold K = kk + 8*(L>=16) + 0..7, then kk + 16 + 8*(L>=16) + 0..7.
__device__ __forceinline__ v16bf load_b_bf16(const unsigned short* src, int ld,
                                             int kk, int N0) {
    int lane = threadIdx.x & 31;
    int g = lane >> 4;
    const unsigned short* r = src + (size_t)(N0 + (lane & 15)) * ld + kk + 8 * g;
    v16us u;
#pragma unroll
    for (int h = 0; h < 16; ++h) u[h] = r[h < 8 ? h : h + 8];
    return __builtin_bit_cast(v16bf, u);
}

// ---------------------------------------------------------------------------
// K1: convert Wq,Wk,Wv,Wo (each 512x512 f32, [out][in]) to bf16 in workspace.
// ---------------------------------------------------------------------------
__global__ void k_cvt_w(const float* __restrict__ Wq, const float* __restrict__ Wk,
                        const float* __restrict__ Wv, const float* __restrict__ Wo,
                        unsigned short* __restrict__ wb) {
    int i = blockIdx.x * 256 + threadIdx.x;     // 0 .. 4*262144-1
    int sel = i >> 18;
    const float* src = sel == 0 ? Wq : sel == 1 ? Wk : sel == 2 ? Wv : Wo;
    wb[i] = f2bf(src[i & 0x3FFFF]);
}

// ---------------------------------------------------------------------------
// K1b: one-pass prep: xb = bf16(x), xpb = bf16(x + pos_emb[s]).
// 4 elements per thread; grid 12800 x 256.
// ---------------------------------------------------------------------------
__global__ void k_prep(const float* __restrict__ x, const float* __restrict__ pos,
                       unsigned short* __restrict__ xb, unsigned short* __restrict__ xpb) {
    size_t i = ((size_t)blockIdx.x * 256 + threadIdx.x) * 4;
    int col = (int)(i & (ND - 1));
    int row = (int)(i >> 9);
    int s   = row % NS;
    const float* xr = x + i;
    const float* pr = pos + (size_t)s * ND + col;
    unsigned short tb[4], tp[4];
#pragma unroll
    for (int j = 0; j < 4; ++j) {
        float xv = xr[j];
        tb[j] = f2bf(xv);
        tp[j] = f2bf(xv + pr[j]);
    }
    ushort4* ob  = (ushort4*)(xb + i);
    ushort4* op  = (ushort4*)(xpb + i);
    *ob = make_ushort4(tb[0], tb[1], tb[2], tb[3]);
    *op = make_ushort4(tp[0], tp[1], tp[2], tp[3]);
}

// ---------------------------------------------------------------------------
// K2: fused QKV projection. C = A_bf16 @ W^T + bias, stored bf16.
// grid (200, 8, 3); block 256 (8 waves). Workgroup stages a 64x512 bf16
// weight panel (64 KB) in LDS; each wave computes a 16x64 strip (4 WMMA
// accumulators -> 4x A-fragment reuse).
// ---------------------------------------------------------------------------
__global__ void k_qkv(const unsigned short* __restrict__ xpb,
                      const unsigned short* __restrict__ xb,
                      const unsigned short* __restrict__ wball,
                      const float* __restrict__ bq, const float* __restrict__ bk,
                      const float* __restrict__ bv,
                      unsigned short* __restrict__ qb, unsigned short* __restrict__ kb,
                      unsigned short* __restrict__ vb) {
    extern __shared__ unsigned short sW[];             // [64][512] bf16 = 64 KB
    int sel = blockIdx.z;                              // 0=Q, 1=K, 2=V
    const unsigned short* wpanel =
        wball + (size_t)sel * ND * ND + (size_t)blockIdx.y * 64 * ND;
    // cooperative 64 KB panel copy (8 B per iteration per thread)
    for (int i = threadIdx.x; i < 64 * ND / 4; i += 256)
        ((uint2*)sW)[i] = ((const uint2*)wpanel)[i];
    __syncthreads();

    int wave = threadIdx.x >> 5;
    int lane = threadIdx.x & 31;
    int g = lane >> 4, ln = lane & 15;
    int M0 = blockIdx.x * 128 + wave * 16;

    const unsigned short* asrc = (sel < 2) ? xpb : xb; // V uses x w/o positions
    const unsigned short* ar = asrc + (size_t)(M0 + ln) * ND;

    v8f c[4] = {};
    for (int kk = 0; kk < ND; kk += 32) {
        __builtin_prefetch(ar + kk + 64, 0, 0);        // next A k-slice
        v16us ua;
#pragma unroll
        for (int v = 0; v < 8; ++v) {
            int k0 = (v < 4 ? 2 * v : 16 + 2 * (v - 4)) + 8 * g + kk;
            ua[2 * v]     = ar[k0];
            ua[2 * v + 1] = ar[k0 + 1];
        }
        v16bf a = __builtin_bit_cast(v16bf, ua);
#pragma unroll
        for (int j = 0; j < 4; ++j) {
            v16bf bm = load_b_bf16(sW, ND, kk, 16 * j);
            c[j] = __builtin_amdgcn_wmma_f32_16x16x32_bf16(false, a, false, bm,
                                                           (short)0, c[j], false, false);
        }
    }

    const float* bias = sel == 0 ? bq : sel == 1 ? bk : bv;
    unsigned short* outb = sel == 0 ? qb : sel == 1 ? kb : vb;
#pragma unroll
    for (int j = 0; j < 4; ++j) {
        int n = blockIdx.y * 64 + 16 * j + ln;
        float bz = bias[n];
#pragma unroll
        for (int i = 0; i < 8; ++i) {
            int rr = M0 + i + 8 * g;                   // C layout: M = i + 8*(lane/16)
            outb[(size_t)rr * ND + n] = f2bf(c[j][i] + bz);
        }
    }
}

// ---------------------------------------------------------------------------
// K3: attention per (b,h). LDS: sQ[112][64] sK[112][64] sVt[64][128]
//     sS[112][128] f32, sP[112][128] bf16.  Total 128 KB dynamic LDS.
// ---------------------------------------------------------------------------
__global__ void k_attn(const unsigned short* __restrict__ qb,
                       const unsigned short* __restrict__ kb,
                       const unsigned short* __restrict__ vb,
                       const float* __restrict__ tdp,
                       float* __restrict__ attn_out,
                       unsigned short* __restrict__ ctxb) {
    extern __shared__ char smem[];
    unsigned short* sQ  = (unsigned short*)(smem);            // 14336 B
    unsigned short* sK  = (unsigned short*)(smem + 14336);    // 14336 B
    unsigned short* sVt = (unsigned short*)(smem + 28672);    // 16384 B
    float*          sS  = (float*)(smem + 45056);             // 57344 B
    unsigned short* sP  = (unsigned short*)(smem + 102400);   // 28672 B

    int b = blockIdx.x >> 3, h = blockIdx.x & 7;
    int t = threadIdx.x;
    float td = *tdp;

    const unsigned short* qg = qb + (size_t)b * NS * ND + h * NHD;
    const unsigned short* kg = kb + (size_t)b * NS * ND + h * NHD;
    const unsigned short* vg = vb + (size_t)b * NS * ND + h * NHD;

    for (int i = t; i < SP * NHD; i += 256) {                 // Q, K zero-padded rows
        int rr = i >> 6, cc = i & 63;
        sQ[i] = rr < NS ? qg[(size_t)rr * ND + cc] : 0;
        sK[i] = rr < NS ? kg[(size_t)rr * ND + cc] : 0;
    }
    for (int i = t; i < NHD * KP; i += 256) {                 // V transposed: [hd][key]
        int hd = i >> 7, key = i & 127;
        sVt[i] = key < NS ? vg[(size_t)key * ND + hd] : 0;
    }
    __syncthreads();

    int wave = t >> 5, lane = t & 31, g = lane >> 4, ln = lane & 15;

    // ---- scores = Q K^T / 8 * decay[key], masked ----
    for (int tile = wave; tile < 49; tile += 8) {
        int M0 = (tile / 7) * 16, N0 = (tile % 7) * 16;
        v8f c = {};
        for (int kk = 0; kk < NHD; kk += 32) {
            v16bf a  = load_a_bf16(sQ, NHD, M0, kk);
            v16bf bm = load_b_bf16(sK, NHD, kk, N0);          // B(d,key) = K[key][d]
            c = __builtin_amdgcn_wmma_f32_16x16x32_bf16(false, a, false, bm,
                                                        (short)0, c, false, false);
        }
        int key = N0 + ln;                                    // constant per lane
        float dec = key < NS ? powf(td, (float)(NS - 1 - key)) : 0.f;
#pragma unroll
        for (int i = 0; i < 8; ++i) {
            int m = M0 + i + 8 * g;
            float v = key < NS ? c[i] * 0.125f * dec : -3.0e38f;
            sS[m * KP + key] = v;
        }
    }
    __syncthreads();

    // ---- softmax over keys (rows handled by threads 0..111) ----
    if (t < SP) {
        float* row = sS + t * KP;
        float mx = -3.0e38f;
        for (int c2 = 0; c2 < NS; ++c2) mx = fmaxf(mx, row[c2]);
        float sum = 0.f;
        for (int c2 = 0; c2 < NS; ++c2) { float e = __expf(row[c2] - mx); row[c2] = e; sum += e; }
        float inv = 1.f / sum;
        unsigned short* prow = sP + t * KP;
        float* ao = attn_out + (((size_t)(b * NH + h)) * NS + (t < NS ? t : 0)) * NS;
        for (int c2 = 0; c2 < NS; ++c2) {
            float p = row[c2] * inv;
            prow[c2] = f2bf(p);
            if (t < NS) ao[c2] = p;                           // f32 attn output
        }
        for (int c2 = NS; c2 < KP; ++c2) prow[c2] = 0;        // pad K dim with zeros
    }
    __syncthreads();

    // ---- ctx = attn @ V ----
    for (int tile = wave; tile < 28; tile += 8) {
        int M0 = (tile >> 2) * 16, N0 = (tile & 3) * 16;
        v8f c = {};
        for (int kk = 0; kk < KP; kk += 32) {
            v16bf a  = load_a_bf16(sP, KP, M0, kk);
            v16bf bm = load_b_bf16(sVt, KP, kk, N0);          // B(key,hd) = Vt[hd][key]
            c = __builtin_amdgcn_wmma_f32_16x16x32_bf16(false, a, false, bm,
                                                        (short)0, c, false, false);
        }
#pragma unroll
        for (int i = 0; i < 8; ++i) {
            int r = M0 + i + 8 * g;
            if (r < NS)
                ctxb[((size_t)(b * NS + r)) * ND + h * NHD + N0 + ln] = f2bf(c[i]);
        }
    }
}

// ---------------------------------------------------------------------------
// K4: out = ctx @ Wo^T + bo (f32 output). Same LDS-panel structure as K2.
// grid (200, 8); 64 KB dynamic LDS.
// ---------------------------------------------------------------------------
__global__ void k_out(const unsigned short* __restrict__ ctxb,
                      const unsigned short* __restrict__ wball,
                      const float* __restrict__ bo, float* __restrict__ out) {
    extern __shared__ unsigned short sW[];             // [64][512] bf16 = 64 KB
    const unsigned short* wpanel =
        wball + (size_t)3 * ND * ND + (size_t)blockIdx.y * 64 * ND;
    for (int i = threadIdx.x; i < 64 * ND / 4; i += 256)
        ((uint2*)sW)[i] = ((const uint2*)wpanel)[i];
    __syncthreads();

    int wave = threadIdx.x >> 5;
    int lane = threadIdx.x & 31;
    int g = lane >> 4, ln = lane & 15;
    int M0 = blockIdx.x * 128 + wave * 16;
    const unsigned short* ar = ctxb + (size_t)(M0 + ln) * ND;

    v8f c[4] = {};
    for (int kk = 0; kk < ND; kk += 32) {
        __builtin_prefetch(ar + kk + 64, 0, 0);
        v16us ua;
#pragma unroll
        for (int v = 0; v < 8; ++v) {
            int k0 = (v < 4 ? 2 * v : 16 + 2 * (v - 4)) + 8 * g + kk;
            ua[2 * v]     = ar[k0];
            ua[2 * v + 1] = ar[k0 + 1];
        }
        v16bf a = __builtin_bit_cast(v16bf, ua);
#pragma unroll
        for (int j = 0; j < 4; ++j) {
            v16bf bm = load_b_bf16(sW, ND, kk, 16 * j);
            c[j] = __builtin_amdgcn_wmma_f32_16x16x32_bf16(false, a, false, bm,
                                                           (short)0, c[j], false, false);
        }
    }
#pragma unroll
    for (int j = 0; j < 4; ++j) {
        int n = blockIdx.y * 64 + 16 * j + ln;
        float bz = bo[n];
#pragma unroll
        for (int i = 0; i < 8; ++i) {
            int r = M0 + i + 8 * g;
            out[(size_t)r * ND + n] = c[j][i] + bz;
        }
    }
}

// ---------------------------------------------------------------------------
extern "C" void kernel_launch(void* const* d_in, const int* in_sizes, int n_in,
                              void* d_out, int out_size, void* d_ws, size_t ws_size,
                              hipStream_t stream) {
    (void)in_sizes; (void)n_in; (void)out_size; (void)ws_size;
    const float* x   = (const float*)d_in[0];
    const float* pos = (const float*)d_in[1];
    const float* td  = (const float*)d_in[2];
    const float* Wq  = (const float*)d_in[3];
    const float* bq  = (const float*)d_in[4];
    const float* Wk  = (const float*)d_in[5];
    const float* bk  = (const float*)d_in[6];
    const float* Wv  = (const float*)d_in[7];
    const float* bv  = (const float*)d_in[8];
    const float* Wo  = (const float*)d_in[9];
    const float* bo  = (const float*)d_in[10];

    float* out  = (float*)d_out;
    float* attn = out + (size_t)NROWS * ND;            // 13,107,200 f32 offset

    // workspace layout (bytes):
    //   [0, 2MB)  weights bf16 (4 x 512x512)
    //   then xpb, xb, Q, K, V, ctx bf16, each 25600*512*2 = 26,214,400 B
    char* ws = (char*)d_ws;
    const size_t WSZ = (size_t)4 * ND * ND * 2;        // 2 MB
    const size_t MSZ = (size_t)NROWS * ND * 2;         // 26,214,400 B
    unsigned short* wb   = (unsigned short*)ws;
    unsigned short* xpb  = (unsigned short*)(ws + WSZ);
    unsigned short* xb   = (unsigned short*)(ws + WSZ + MSZ);
    unsigned short* qb   = (unsigned short*)(ws + WSZ + 2 * MSZ);
    unsigned short* kb   = (unsigned short*)(ws + WSZ + 3 * MSZ);
    unsigned short* vb   = (unsigned short*)(ws + WSZ + 4 * MSZ);
    unsigned short* ctxb = (unsigned short*)(ws + WSZ + 5 * MSZ);

    k_cvt_w<<<dim3(4096), dim3(256), 0, stream>>>(Wq, Wk, Wv, Wo, wb);
    k_prep<<<dim3(12800), dim3(256), 0, stream>>>(x, pos, xb, xpb);
    k_qkv<<<dim3(200, 8, 3), dim3(256), 65536, stream>>>(xpb, xb, wb, bq, bk, bv, qb, kb, vb);
    k_attn<<<dim3(NB * NH), dim3(256), 131072, stream>>>(qb, kb, vb, td, attn, ctxb);
    k_out<<<dim3(200, 8), dim3(256), 65536, stream>>>(ctxb, wb, bo, out);
}
